// G_synthesis_co_mod_gan_48344151884071
// MI455X (gfx1250) — matrix operand order repaired and checked by reading, not compiled, not run on hardware
//
#include <hip/hip_runtime.h>

typedef __attribute__((ext_vector_type(16))) __bf16 v16bf;
typedef __attribute__((ext_vector_type(8)))  __bf16 v8bf;
typedef __attribute__((ext_vector_type(8)))  float  v8f;

#define B_    4
#define CIN_  512
#define COUT_ 512
#define SD_   512
#define H_    64
#define HO_   129   // pre-blur spatial size
#define HF_   128   // final spatial size

// ---------------- small prep kernels ----------------

// s[b][cin] = style[b] . mod_w[cin] * (1/sqrt(SD)) + mod_b[cin]
__global__ void k_style(const float* __restrict__ style, const float* __restrict__ mod_w,
                        const float* __restrict__ mod_b, float* __restrict__ s) {
  int i = blockIdx.x * blockDim.x + threadIdx.x;
  if (i >= B_ * CIN_) return;
  int b = i / CIN_, cin = i % CIN_;
  const float lin = 0.04419417382415922f; // 1/sqrt(512)
  const float* st = style + (size_t)b * SD_;
  const float* mw = mod_w + (size_t)cin * SD_;
  float acc = 0.f;
  for (int k = 0; k < SD_; ++k) acc += st[k] * mw[k];
  s[i] = acc * lin + mod_b[cin];
}

// wsq[cout][cin] = sum over 3x3 of weight^2
__global__ void k_wsq(const float* __restrict__ w, float* __restrict__ wsq) {
  int i = blockIdx.x * blockDim.x + threadIdx.x;
  if (i >= COUT_ * CIN_) return;
  const float* p = w + (size_t)i * 9;
  float a = 0.f;
#pragma unroll
  for (int t = 0; t < 9; ++t) a += p[t] * p[t];
  wsq[i] = a;
}

// demod[b][cout] = rsqrt(scale^2 * sum_cin s[b,cin]^2 * wsq[cout,cin] + 1e-8)
__global__ void k_demod(const float* __restrict__ s, const float* __restrict__ wsq,
                        float* __restrict__ demod) {
  int i = blockIdx.x * blockDim.x + threadIdx.x;
  if (i >= B_ * COUT_) return;
  int b = i / COUT_, co = i % COUT_;
  const float scale2 = 1.0f / 4608.0f; // (1/sqrt(CIN*9))^2
  const float* sv = s + (size_t)b * CIN_;
  const float* wq = wsq + (size_t)co * CIN_;
  float a = 0.f;
  for (int c = 0; c < CIN_; ++c) { float sc = sv[c]; a += sc * sc * wq[c]; }
  demod[i] = rsqrtf(scale2 * a + 1e-8f);
}

// xb[b][y][x][cin] = bf16(x[b][cin][y][x])   (NHWC, cin contiguous for B fragments)
__global__ void k_pack_x(const float* __restrict__ x, __bf16* __restrict__ xb) {
  int i = blockIdx.x * blockDim.x + threadIdx.x;
  if (i >= B_ * CIN_ * H_ * H_) return;
  int xx = i % H_;
  int yy = (i / H_) % H_;
  int cin = (i / (H_ * H_)) % CIN_;
  int b = i / (CIN_ * H_ * H_);
  float v = x[(((size_t)b * CIN_ + cin) * H_ + yy) * H_ + xx];
  xb[(((size_t)b * H_ + yy) * H_ + xx) * CIN_ + cin] = (__bf16)v;
}

// wmod[b][ky][kx][cout][cin] = bf16(scale * weight[cout][cin][ky][kx] * s[b][cin] * demod[b][cout])
__global__ void k_pack_w(const float* __restrict__ weight, const float* __restrict__ s,
                         const float* __restrict__ demod, __bf16* __restrict__ wmod) {
  int i = blockIdx.x * blockDim.x + threadIdx.x;
  if (i >= B_ * 9 * COUT_ * CIN_) return;
  int cin  = i % CIN_;
  int cout = (i / CIN_) % COUT_;
  int t    = (i / (CIN_ * COUT_)) % 9;
  int b    = i / (9 * CIN_ * COUT_);
  const float scale = 0.01473139127471974f; // 1/sqrt(4608)
  float v = scale * weight[((size_t)cout * CIN_ + cin) * 9 + t]
                  * s[(size_t)b * CIN_ + cin]
                  * demod[(size_t)b * COUT_ + cout];
  wmod[((((size_t)b * 9) + t) * COUT_ + cout) * CIN_ + cin] = (__bf16)v;
}

// ---------------- main WMMA implicit-GEMM transposed conv ----------------
// One wave (32 threads) computes a 64(cout) x 16(same-parity ox pixels) f32 tile
// of the 129x129 pre-blur output for one (b, oy) row: 4 accumulators sharing
// each B fragment (4x operand reuse, 4 independent WMMAs per K-step to cover
// the XDL dependency window), over up to 4 taps x (CIN/32) K-steps.
__global__ void __launch_bounds__(32)
k_tconv_wmma(const __bf16* __restrict__ xb, const __bf16* __restrict__ wmod,
             float* __restrict__ tmp) {
  const int tile  = blockIdx.x % 9;
  const int oy    = blockIdx.x / 9;
  const int cout0 = blockIdx.y * 64;
  const int b     = blockIdx.z;

  int p, ox0;
  if (tile < 5) { p = 0; ox0 = tile * 32; }           // even-ox tiles: ox = ox0 + 2n
  else          { p = 1; ox0 = 1 + (tile - 5) * 32; } // odd-ox tiles

  const int lane = threadIdx.x;
  const int n    = lane & 15;   // N index (pixel) / M index (cout row within sub-tile)
  const int h    = lane >> 4;   // K-half selector per ISA fragment layout

  v8f c[4];
#pragma unroll
  for (int j = 0; j < 4; ++j)
#pragma unroll
    for (int q = 0; q < 8; ++q) c[j][q] = 0.0f;

#pragma unroll
  for (int ky = 0; ky < 3; ++ky) {
    if ((oy + ky) & 1) continue;            // dilated-row parity
    int iy = (oy + ky - 2) >> 1;            // exact: oy+ky-2 is even
    if (iy < 0 || iy >= H_) continue;
#pragma unroll
    for (int kx = 0; kx < 3; ++kx) {
      if ((p + kx) & 1) continue;           // dilated-col parity
      int ixbase = (ox0 + kx - 2) >> 1;     // exact: even
      if (ixbase >= H_ || ixbase + 15 < 0) continue;
      int ix = ixbase + n;
      bool bvalid = (ix >= 0) && (ix < H_);

      // flipped kernel tap: wf[ky][kx] = w[2-ky][2-kx]
      const __bf16* wrow = wmod
          + ((((size_t)b * 9) + (size_t)(2 - ky) * 3 + (2 - kx)) * COUT_ + (cout0 + n)) * CIN_;
      const __bf16* xrow = xb + (((size_t)b * H_ + iy) * H_ + (bvalid ? ix : 0)) * CIN_;

      for (int k0 = 0; k0 < CIN_; k0 += 32) {
        // B fragment (32x16 bf16): lane holds 16 contiguous K for pixel n
        v16bf bb;
#pragma unroll
        for (int q = 0; q < 16; ++q) bb[q] = (__bf16)0.0f;
        if (bvalid) bb = *reinterpret_cast<const v16bf*>(xrow + k0 + 16 * h);

        // 4 cout sub-tiles reuse bb; independent WMMAs hide XDL latency
#pragma unroll
        for (int j = 0; j < 4; ++j) {
          const __bf16* wj = wrow + (size_t)j * 16 * CIN_ + k0;
          // A fragment (16x32 bf16): lane h=0 -> K 0..15 ; h=1 -> K 16..31
          v8bf lo = *reinterpret_cast<const v8bf*>(wj + 16 * h);
          v8bf hi = *reinterpret_cast<const v8bf*>(wj + 16 * h + 8);
          v16bf a;
#pragma unroll
          for (int q = 0; q < 8; ++q) { a[q] = lo[q]; a[q + 8] = hi[q]; }

          c[j] = __builtin_amdgcn_wmma_f32_16x16x32_bf16(
                     false, a, false, bb, (short)0, c[j], false, false);
        }
      }
    }
  }

  // C/D layout: lane -> column N = lane&15; VGPR v -> row M = v + 8*(lane>>4)
  int ox = ox0 + 2 * n;
  if (ox < HO_) {
#pragma unroll
    for (int j = 0; j < 4; ++j) {
      size_t base = (((size_t)b * COUT_ + cout0 + j * 16 + h * 8) * HO_ + oy) * HO_ + ox;
#pragma unroll
      for (int v = 0; v < 8; ++v) tmp[base + (size_t)v * HO_ * HO_] = c[j][v];
    }
  }
}

// ---------------- blur + noise + bias + leaky-relu epilogue ----------------
__global__ void k_blur(const float* __restrict__ tmp, const float* __restrict__ noise,
                       const float* __restrict__ noise_w, const float* __restrict__ act_bias,
                       float* __restrict__ out) {
  int i = blockIdx.x * blockDim.x + threadIdx.x;
  if (i >= B_ * COUT_ * HF_ * HF_) return;
  int ox = i % HF_;
  int oy = (i / HF_) % HF_;
  int c  = (i / (HF_ * HF_)) % COUT_;
  int b  = i / (COUT_ * HF_ * HF_);
  const float cw[4] = {0.25f, 0.75f, 0.75f, 0.25f}; // separable [1,3,3,1]/4
  const float* tb = tmp + ((size_t)b * COUT_ + c) * HO_ * HO_;
  float acc = 0.f;
#pragma unroll
  for (int dy = 0; dy < 4; ++dy) {
    int ty = oy + dy - 1;
    if ((unsigned)ty >= (unsigned)HO_) continue;
    float r = 0.f;
#pragma unroll
    for (int dx = 0; dx < 4; ++dx) {
      int tx = ox + dx - 1;
      if ((unsigned)tx >= (unsigned)HO_) continue;
      r += cw[dx] * tb[(size_t)ty * HO_ + tx];
    }
    acc += cw[dy] * r;
  }
  acc += noise_w[0] * noise[((size_t)b * HF_ + oy) * HF_ + ox];
  float v = acc + act_bias[c];
  v = (v > 0.f ? v : 0.2f * v) * 1.4142135623730951f;
  out[i] = v;
}

// ---------------- launch ----------------
extern "C" void kernel_launch(void* const* d_in, const int* in_sizes, int n_in,
                              void* d_out, int out_size, void* d_ws, size_t ws_size,
                              hipStream_t stream) {
  const float* x       = (const float*)d_in[0];
  const float* style   = (const float*)d_in[1];
  const float* noise   = (const float*)d_in[2];
  const float* weight  = (const float*)d_in[3];
  const float* mod_w   = (const float*)d_in[4];
  const float* mod_b   = (const float*)d_in[5];
  const float* noise_w = (const float*)d_in[6];
  const float* act_b   = (const float*)d_in[7];
  float* out = (float*)d_out;

  char* ws = (char*)d_ws;
  size_t off = 0;
  float*  s_buf   = (float*)(ws + off); off += (size_t)B_ * CIN_ * sizeof(float);
  float*  demod   = (float*)(ws + off); off += (size_t)B_ * COUT_ * sizeof(float);
  float*  wsq     = (float*)(ws + off); off += (size_t)COUT_ * CIN_ * sizeof(float);
  __bf16* xb      = (__bf16*)(ws + off); off += (size_t)B_ * H_ * H_ * CIN_ * sizeof(__bf16);
  __bf16* wmod    = (__bf16*)(ws + off); off += (size_t)B_ * 9 * COUT_ * CIN_ * sizeof(__bf16);
  float*  tmp     = (float*)(ws + off); off += (size_t)B_ * COUT_ * HO_ * HO_ * sizeof(float);

  k_style<<<(B_ * CIN_ + 255) / 256, 256, 0, stream>>>(style, mod_w, mod_b, s_buf);
  k_wsq<<<(COUT_ * CIN_ + 255) / 256, 256, 0, stream>>>(weight, wsq);
  k_demod<<<(B_ * COUT_ + 255) / 256, 256, 0, stream>>>(s_buf, wsq, demod);
  k_pack_x<<<(B_ * CIN_ * H_ * H_ + 255) / 256, 256, 0, stream>>>(x, xb);
  k_pack_w<<<(B_ * 9 * COUT_ * CIN_ + 255) / 256, 256, 0, stream>>>(weight, s_buf, demod, wmod);

  dim3 grid(9 * HO_, COUT_ / 64, B_);
  k_tconv_wmma<<<grid, 32, 0, stream>>>(xb, wmod, tmp);

  k_blur<<<((size_t)B_ * COUT_ * HF_ * HF_ + 255) / 256, 256, 0, stream>>>(
      tmp, noise, noise_w, act_b, out);
}
